// SimpleLSTM_28733331210305
// MI455X (gfx1250) — matrix operand order repaired and checked by reading, not compiled, not run on hardware
//
#include <hip/hip_runtime.h>
#include <hip/hip_bf16.h>

typedef _Float16 v4h  __attribute__((ext_vector_type(4)));
typedef _Float16 v8h  __attribute__((ext_vector_type(8)));
typedef _Float16 v16h __attribute__((ext_vector_type(16)));
typedef float    v8f  __attribute__((ext_vector_type(8)));

#define T_STEPS 512
#define BATCH   128
#define INPUT   256
#define HIDDEN  512
#define GATES   2048           // 4*HIDDEN
#define KTOT    768            // INPUT + HIDDEN
#define OUTPUT  256
#define NBLK    32             // persistent workgroups
#define LDS_PAD_K 776          // 768 + 8 halves -> bank-conflict-minimal stride

__device__ __forceinline__ float sigmoidf_(float x) {
    return 1.0f / (1.0f + __expf(-x));
}
__device__ __forceinline__ float tanhf_(float x) {
    x = fminf(fmaxf(x, -15.0f), 15.0f);
    float e = __expf(2.0f * x);
    return (e - 1.0f) / (e + 1.0f);
}

// A fragment (16x32 f16): lane holds row lane%16; K in {0-7,16-23} (lanes<16)
// or {8-15,24-31} (lanes>=16). p = row base + kbase (+koffA folded by caller).
__device__ __forceinline__ v16h load_afrag(const _Float16* p) {
    v8h lo = *(const v8h*)(p);
    v8h hi = *(const v8h*)(p + 16);
    v16h r;
#pragma unroll
    for (int i = 0; i < 8; ++i) { r[i] = lo[i]; r[8 + i] = hi[i]; }
    return r;
}

// B fragment (32x16 f16): lane holds col lane%16; K in {0-7,16-23} (lanes<16)
// or {8-15,24-31} (lanes>=16). p = col base + kbase + koffB.
__device__ __forceinline__ v16h load_bfrag(const _Float16* p) {
    v8h lo = *(const v8h*)(p);
    v8h hi = *(const v8h*)(p + 16);
    v16h r;
#pragma unroll
    for (int i = 0; i < 8; ++i) { r[i] = lo[i]; r[8 + i] = hi[i]; }
    return r;
}

// monotonic-counter grid barrier (all blocks resident; persistent kernel)
__device__ __forceinline__ void grid_barrier(unsigned* bar, unsigned& epoch) {
    __threadfence();                 // release this thread's stores to device scope
    __syncthreads();
    if (threadIdx.x == 0) {
        ++epoch;
        __hip_atomic_fetch_add(bar, 1u, __ATOMIC_RELEASE, __HIP_MEMORY_SCOPE_AGENT);
        while (__hip_atomic_load(bar, __ATOMIC_ACQUIRE, __HIP_MEMORY_SCOPE_AGENT) <
               epoch * (unsigned)NBLK) {
            __builtin_amdgcn_s_sleep(1);
        }
    }
    __syncthreads();                 // acquire for whole workgroup
}

// ---------------- prep: f16 weights, fused bias, zero state ----------------
__global__ void lstm_prep_kernel(const float* __restrict__ Wih,
                                 const float* __restrict__ Whh,
                                 const float* __restrict__ bih,
                                 const float* __restrict__ bhh,
                                 const float* __restrict__ Wout,
                                 _Float16* __restrict__ Wc,      // [2048][768]
                                 _Float16* __restrict__ Wout16,  // [256][512]
                                 float*    __restrict__ bias,    // [2048]
                                 float*    __restrict__ cbuf,    // [128*512]
                                 _Float16* __restrict__ h0,      // [128*512]
                                 unsigned* __restrict__ bar) {
    int i = blockIdx.x * 256 + threadIdx.x;
    if (i < GATES * KTOT) {
        int g = i / KTOT, k = i - g * KTOT;
        float w = (k < INPUT) ? Wih[g * INPUT + k] : Whh[g * HIDDEN + (k - INPUT)];
        Wc[i] = (_Float16)w;
    }
    if (i < OUTPUT * HIDDEN) Wout16[i] = (_Float16)Wout[i];
    if (i < GATES)           bias[i]   = bih[i] + bhh[i];
    if (i < BATCH * HIDDEN) { cbuf[i] = 0.0f; h0[i] = (_Float16)0.0f; }
    if (i < 64)              bar[i] = 0u;
}

// ---------------- x -> f16 ----------------
__global__ void convx_kernel(const float* __restrict__ x, _Float16* __restrict__ x16) {
    size_t i = ((size_t)blockIdx.x * 256 + threadIdx.x) * 4;
    float4 v = *(const float4*)(x + i);
    v4h o;
    o[0] = (_Float16)v.x; o[1] = (_Float16)v.y;
    o[2] = (_Float16)v.z; o[3] = (_Float16)v.w;
    *(v4h*)(x16 + i) = o;
}

// ---------------- persistent recurrent kernel ----------------
// 32 WGs x 256 threads (8 waves). WG w owns gate columns [64w, 64w+64),
// weights staged in LDS once. Waves tile 4x2: wave = (rw 0..3) x (cw 0..1);
// rows [32rw, 32rw+32), cols [64wg + 32cw, +32). Per K-chunk: 2 A-frags +
// 2 B-frags -> 4 WMMAs (0.5 LDS fragment per WMMA, 2x reuse of each frag).
__global__ void lstm_step_kernel(const _Float16* __restrict__ x16,   // [T][B][256]
                                 const _Float16* __restrict__ Wc,    // [2048][768]
                                 const float*    __restrict__ bias,  // [2048]
                                 float*          __restrict__ gates, // [128][2048]
                                 float*          __restrict__ cbuf,  // [128][512]
                                 const _Float16* __restrict__ h0,    // [128][512], contiguous before hs
                                 _Float16*       __restrict__ hs,    // [T][128][512]
                                 unsigned*       __restrict__ bar) {
    extern __shared__ _Float16 Wlds[];   // [64][LDS_PAD_K]
    const int tid  = threadIdx.x;
    const int wg   = blockIdx.x;
    const int lane = tid & 31;
    const int wave = tid >> 5;

    // stage this WG's 64 weight columns (constant across all timesteps)
    for (int i = tid; i < 64 * KTOT; i += 256) {
        int col = i / KTOT, k = i - col * KTOT;
        Wlds[col * LDS_PAD_K + k] = Wc[(size_t)(wg * 64 + col) * KTOT + k];
    }
    __syncthreads();

    const int n     = lane & 15;
    const int koffA = (lane < 16) ? 0 : 8;
    const int koffB = (lane < 16) ? 0 : 8;
    const int rw    = wave >> 1;            // 0..3 : row group (32 rows)
    const int cw    = wave & 1;             // 0..1 : col group (32 cols)
    const int rowbase = rw * 32;
    const int arow0   = rowbase + n;        // A row, tile 0
    // per-lane LDS bases (halves); col-group term lives in the address VGPR,
    // so all per-instruction DS offsets are small immediates
    const int bbase0 = (cw * 32 + n) * LDS_PAD_K + koffB;
    const int bbase1 = bbase0 + 16 * LDS_PAD_K;

    float breg0 = bias[wg * 64 + cw * 32 + n];
    float breg1 = bias[wg * 64 + cw * 32 + 16 + n];

    // phase-2 element assignment (fixed per thread -> c is thread-private)
    const int e0 = (wg * 256 + tid) * 8;
    const int eb = e0 >> 9;            // batch row
    const int ej = e0 & 511;           // hidden index

    unsigned epoch = 0;
    for (int t = 0; t < T_STEPS; ++t) {
        // ---------- phase 1: gates = [x_t | h_{t-1}] * Wc^T + bias ----------
        // h0 sits contiguously before hs, so hsrc = h0 + t*B*H covers t==0 too.
        const _Float16* hsrc = h0 + (size_t)t * BATCH * HIDDEN;
        const _Float16* axp0 = x16 + (size_t)t * BATCH * INPUT
                                   + (size_t)arow0 * INPUT + koffA;
        const _Float16* axp1 = axp0 + 16 * INPUT;
        const _Float16* ahp0 = hsrc + (size_t)arow0 * HIDDEN + koffA;
        const _Float16* ahp1 = ahp0 + 16 * HIDDEN;

        v8f acc00, acc01, acc10, acc11;
#pragma unroll
        for (int i = 0; i < 8; ++i) {
            acc00[i] = breg0; acc01[i] = breg1;
            acc10[i] = breg0; acc11[i] = breg1;
        }

        // K 0..255 : x_t part (8 chunks of 32)
#pragma unroll
        for (int kb = 0; kb < 8; ++kb) {
            v16h a0 = load_afrag(axp0 + kb * 32);
            v16h a1 = load_afrag(axp1 + kb * 32);
            v16h b0 = load_bfrag(&Wlds[bbase0 + kb * 32]);
            v16h b1 = load_bfrag(&Wlds[bbase1 + kb * 32]);
            acc00 = __builtin_amdgcn_wmma_f32_16x16x32_f16(false, a0, false, b0, (short)0, acc00, false, false);
            acc01 = __builtin_amdgcn_wmma_f32_16x16x32_f16(false, a0, false, b1, (short)0, acc01, false, false);
            acc10 = __builtin_amdgcn_wmma_f32_16x16x32_f16(false, a1, false, b0, (short)0, acc10, false, false);
            acc11 = __builtin_amdgcn_wmma_f32_16x16x32_f16(false, a1, false, b1, (short)0, acc11, false, false);
        }
        // K 256..767 : h_{t-1} part (16 chunks of 32)
#pragma unroll
        for (int kb = 0; kb < 16; ++kb) {
            v16h a0 = load_afrag(ahp0 + kb * 32);
            v16h a1 = load_afrag(ahp1 + kb * 32);
            v16h b0 = load_bfrag(&Wlds[bbase0 + (8 + kb) * 32]);
            v16h b1 = load_bfrag(&Wlds[bbase1 + (8 + kb) * 32]);
            acc00 = __builtin_amdgcn_wmma_f32_16x16x32_f16(false, a0, false, b0, (short)0, acc00, false, false);
            acc01 = __builtin_amdgcn_wmma_f32_16x16x32_f16(false, a0, false, b1, (short)0, acc01, false, false);
            acc10 = __builtin_amdgcn_wmma_f32_16x16x32_f16(false, a1, false, b0, (short)0, acc10, false, false);
            acc11 = __builtin_amdgcn_wmma_f32_16x16x32_f16(false, a1, false, b1, (short)0, acc11, false, false);
        }

        // store gate tiles (C layout: vgpr r -> row r / r+8 by lane half)
        {
            const int srow0 = rowbase + ((lane < 16) ? 0 : 8);
            const int col0  = wg * 64 + cw * 32 + n;
            float* g00 = gates + (size_t)srow0 * GATES + col0;
#pragma unroll
            for (int r = 0; r < 8; ++r) {
                g00[(size_t)r * GATES]                 = acc00[r];
                g00[(size_t)r * GATES + 16]            = acc01[r];
                g00[(size_t)(r + 16) * GATES]          = acc10[r];
                g00[(size_t)(r + 16) * GATES + 16]     = acc11[r];
            }
        }
        grid_barrier(bar, epoch);

        // ---------- phase 2: elementwise LSTM cell ----------
        {
            const float* grow = gates + (size_t)eb * GATES;
            float*    cp = cbuf + (size_t)eb * HIDDEN + ej;
            _Float16* hp = hs + (size_t)t * BATCH * HIDDEN + (size_t)eb * HIDDEN + ej;
#pragma unroll
            for (int u = 0; u < 8; ++u) {
                float iv = sigmoidf_(grow[ej + u]);
                float fv = sigmoidf_(grow[HIDDEN + ej + u]);
                float gv = tanhf_   (grow[2 * HIDDEN + ej + u]);
                float ov = sigmoidf_(grow[3 * HIDDEN + ej + u]);
                float cn = fv * cp[u] + iv * gv;
                cp[u] = cn;
                hp[u] = (_Float16)(ov * tanhf_(cn));
            }
        }
        grid_barrier(bar, epoch);
    }
}

// ---------------- output projection: out = hs @ Wout^T + b_out ----------------
// one 16x16 tile per wave; K = 512 -> 16 WMMA ops
__global__ void outproj_kernel(const _Float16* __restrict__ hs,     // [65536][512]
                               const _Float16* __restrict__ Wout16, // [256][512]
                               const float*    __restrict__ b_out,  // [256]
                               float*          __restrict__ out) {  // [65536][256]
    const int tid  = threadIdx.x;
    const int lane = tid & 31;
    const int wave = tid >> 5;
    const int tile = blockIdx.x * 8 + wave;      // 65536 tiles total
    const int rowTile = tile >> 4;
    const int colTile = tile & 15;

    const int n     = lane & 15;
    const int koffA = (lane < 16) ? 0 : 8;
    const int koffB = (lane < 16) ? 0 : 8;
    const int col   = colTile * 16 + n;

    const _Float16* ap = hs + (size_t)(rowTile * 16 + n) * HIDDEN + koffA;
    const _Float16* bp = Wout16 + (size_t)col * HIDDEN + koffB;

    float bv = b_out[col];
    v8f acc;
#pragma unroll
    for (int i = 0; i < 8; ++i) acc[i] = bv;

#pragma unroll
    for (int kb = 0; kb < 16; ++kb) {
        v16h a = load_afrag(ap + kb * 32);
        v16h b = load_bfrag(bp + kb * 32);
        acc = __builtin_amdgcn_wmma_f32_16x16x32_f16(
            false, a, false, b, (short)0, acc, false, false);
    }
    const int srow = rowTile * 16 + ((lane < 16) ? 0 : 8);
    float* op = out + (size_t)srow * OUTPUT + col;
#pragma unroll
    for (int r = 0; r < 8; ++r) op[(size_t)r * OUTPUT] = acc[r];
}

extern "C" void kernel_launch(void* const* d_in, const int* in_sizes, int n_in,
                              void* d_out, int out_size, void* d_ws, size_t ws_size,
                              hipStream_t stream) {
    const float* x    = (const float*)d_in[0];   // [512,128,256]
    const float* Wih  = (const float*)d_in[1];   // [2048,256]
    const float* Whh  = (const float*)d_in[2];   // [2048,512]
    const float* bih  = (const float*)d_in[3];   // [2048]
    const float* bhh  = (const float*)d_in[4];   // [2048]
    const float* Wout = (const float*)d_in[5];   // [256,512]
    const float* bout = (const float*)d_in[6];   // [256]
    float* out = (float*)d_out;                  // [512,128,256]

    char* ws = (char*)d_ws;
    // workspace layout (all offsets 256B aligned); h0 MUST directly precede hs
    _Float16* x16    = (_Float16*)(ws + 0);                       // 33,554,432 B
    _Float16* Wc     = (_Float16*)(ws + 33554432);                //  3,145,728 B
    _Float16* Wout16 = (_Float16*)(ws + 36700160);                //    262,144 B
    float*    bias   = (float*)   (ws + 36962304);                //      8,192 B
    float*    gates  = (float*)   (ws + 36970496);                //  1,048,576 B
    float*    cbuf   = (float*)   (ws + 38019072);                //    262,144 B
    unsigned* bar    = (unsigned*)(ws + 38281216);                //        256 B
    _Float16* h0     = (_Float16*)(ws + 38281472);                //    131,072 B
    _Float16* hs     = (_Float16*)(ws + 38412544);                // 67,108,864 B
    // total ~105.5 MB

    // 1) prep: f16 weights, fused bias, zero c/h0/barrier
    lstm_prep_kernel<<<(GATES * KTOT + 255) / 256, 256, 0, stream>>>(
        Wih, Whh, bih, bhh, Wout, Wc, Wout16, bias, cbuf, h0, bar);

    // 2) x -> f16 (16,777,216 elements, 4/thread)
    convx_kernel<<<16384, 256, 0, stream>>>(x, x16);

    // 3) persistent recurrent kernel: 32 WGs, 97 KB LDS each (weights resident)
    lstm_step_kernel<<<NBLK, 256, 64 * LDS_PAD_K * sizeof(_Float16), stream>>>(
        x16, Wc, bias, gates, cbuf, h0, hs, bar);

    // 4) output projection: 65536 tiles / 8 waves per block
    outproj_kernel<<<8192, 256, 0, stream>>>(hs, Wout16, bout, out);
}
